// FeaturesLoss_6227702579594
// MI455X (gfx1250) — matrix operand order repaired and checked by reading, not compiled, not run on hardware
//
#include <hip/hip_runtime.h>
#include <hip/hip_bf16.h>

typedef __bf16 bf16_t;
typedef __attribute__((ext_vector_type(16))) __bf16 v16bf;
typedef __attribute__((ext_vector_type(8)))  float  v8f;

#define N_ROWS 8192
#define DIM    512
#define PD_EPS 1e-6f
// D * eps^2 = 512 * 1e-12
#define DEPS2  5.12e-10f

union FragB16 { v16bf v; uint4 q[2]; };
union AccF32  { v8f v; float f[8]; };

// ---------------------------------------------------------------------------
// Kernel 0: zero the class-count histogram (deterministic re-init every call)
// ---------------------------------------------------------------------------
__global__ __launch_bounds__(64) void init_kernel(int* cnt) {
    if (threadIdx.x < 16) cnt[threadIdx.x] = 0;
}

// ---------------------------------------------------------------------------
// Kernel 1: row L2-normalize (exact f32), emit bf16 hi/lo split, r, s, counts.
// One wave32 per row; lane owns 16 contiguous floats (64B -> 4x float4).
// ---------------------------------------------------------------------------
__global__ __launch_bounds__(256) void rownorm_kernel(
    const float* __restrict__ emb, const int* __restrict__ label,
    bf16_t* __restrict__ hi, bf16_t* __restrict__ lo,
    float* __restrict__ r, float* __restrict__ s, int* __restrict__ cnt)
{
    const int wave = threadIdx.x >> 5;
    const int lane = threadIdx.x & 31;
    const int row  = blockIdx.x * 8 + wave;

    const float* src = emb + (size_t)row * DIM + lane * 16;
    float4 x0 = ((const float4*)src)[0];
    float4 x1 = ((const float4*)src)[1];
    float4 x2 = ((const float4*)src)[2];
    float4 x3 = ((const float4*)src)[3];

    float vals[16] = { x0.x,x0.y,x0.z,x0.w, x1.x,x1.y,x1.z,x1.w,
                       x2.x,x2.y,x2.z,x2.w, x3.x,x3.y,x3.z,x3.w };

    float ss = 0.0f;
#pragma unroll
    for (int e = 0; e < 16; ++e) ss += vals[e] * vals[e];
#pragma unroll
    for (int off = 16; off > 0; off >>= 1) ss += __shfl_xor(ss, off, 32);

    const float inv = 1.0f / fmaxf(sqrtf(ss), 1e-12f);

    FragB16 H, L;
    float rp = 0.0f, sp = 0.0f;
#pragma unroll
    for (int e = 0; e < 16; ++e) {
        float ne = vals[e] * inv;
        bf16_t h = (bf16_t)ne;            // RNE truncation to bf16
        float  hf = (float)h;
        bf16_t l = (bf16_t)(ne - hf);     // residual
        H.v[e] = h;
        L.v[e] = l;
        rp += ne * ne;
        sp += ne;
    }
#pragma unroll
    for (int off = 16; off > 0; off >>= 1) {
        rp += __shfl_xor(rp, off, 32);
        sp += __shfl_xor(sp, off, 32);
    }

    uint4* dh = (uint4*)(hi + (size_t)row * DIM + lane * 16);
    uint4* dl = (uint4*)(lo + (size_t)row * DIM + lane * 16);
    dh[0] = H.q[0]; dh[1] = H.q[1];
    dl[0] = L.q[0]; dl[1] = L.q[1];

    if (lane == 0) {
        r[row] = rp;
        s[row] = sp;
        atomicAdd(&cnt[label[row] & 15], 1);
    }
}

// ---------------------------------------------------------------------------
// Kernel 2: exact integer mask count -> 1/pos_den, 1/neg_den
// ---------------------------------------------------------------------------
__global__ __launch_bounds__(32) void denom_kernel(const int* __restrict__ cnt,
                                                   float* __restrict__ dens)
{
    if (threadIdx.x == 0) {
        long long pos = 0;
        for (int k = 0; k < 10; ++k) pos += (long long)cnt[k] * (long long)cnt[k];
        double posd = (double)pos;
        double tot  = (double)N_ROWS * (double)N_ROWS;
        dens[0] = (float)(1.0 / (2.0 * posd + 1e-7));
        dens[1] = (float)(1.0 / (2.0 * (tot - posd) + 1e-7));
    }
}

// ---------------------------------------------------------------------------
// Fragment loads per ISA 7.12.2 (16-bit, wave32).
// A 16x32: lane = half*16 + m; VGPRs 0-3 hold K = half*8 + {0..7},
//          VGPRs 4-7 hold K = 16 + half*8 + {0..7}  -> two 16B chunks.
// B 32x16: lane = half*16 + n; same K split as A (half*8 + {0..7}, 16 + half*8 + {0..7}).
// ---------------------------------------------------------------------------
__device__ inline v16bf load_afrag(const bf16_t* __restrict__ mat,
                                   size_t rowbase, int k0, int half)
{
    FragB16 f;
    const bf16_t* p = mat + rowbase + k0 + half * 8;
    f.q[0] = *(const uint4*)(p);        // K = k0 + half*8 .. +7
    f.q[1] = *(const uint4*)(p + 16);   // K = k0 + 16 + half*8 .. +7
    return f.v;
}

__device__ inline v16bf load_bfrag(const bf16_t* __restrict__ mat,
                                   size_t rowbase, int k0, int half)
{
    FragB16 f;
    const bf16_t* p = mat + rowbase + k0 + half * 8;
    f.q[0] = *(const uint4*)(p);        // K = k0 + half*8 .. +7
    f.q[1] = *(const uint4*)(p + 16);   // K = k0 + 16 + half*8 .. +7
    return f.v;
}

// ---------------------------------------------------------------------------
// Kernel 3: fused bf16x3 gram GEMM + contrastive-loss epilogue.
// Grid (64, 128): WG tile = 128 rows x 64 cols. 8 waves stacked in M;
// each wave -> 16(M) x 64(N): 4 f32 accumulators, A-fragment reused 4x.
// k-loop: 16 steps x (3 wmma x 4 tiles) = 192 v_wmma_f32_16x16x32_bf16 / wave.
// ---------------------------------------------------------------------------
__global__ __launch_bounds__(256) void gemm_loss_kernel(
    const bf16_t* __restrict__ hi, const bf16_t* __restrict__ lo,
    const float* __restrict__ r, const float* __restrict__ s,
    const int* __restrict__ label, const float* __restrict__ dens,
    float* __restrict__ partials)
{
    __shared__ float wsum[8];

    const int wave = threadIdx.x >> 5;
    const int lane = threadIdx.x & 31;
    const int half = lane >> 4;
    const int lq   = lane & 15;

    const int m0 = blockIdx.x * 128 + wave * 16;
    const int n0 = blockIdx.y * 64;

    v8f acc0 = {}, acc1 = {}, acc2 = {}, acc3 = {};

    const size_t abase  = (size_t)(m0 + lq) * DIM;
    const size_t bbase0 = (size_t)(n0 +  0 + lq) * DIM;
    const size_t bbase1 = (size_t)(n0 + 16 + lq) * DIM;
    const size_t bbase2 = (size_t)(n0 + 32 + lq) * DIM;
    const size_t bbase3 = (size_t)(n0 + 48 + lq) * DIM;

    for (int k0 = 0; k0 < DIM; k0 += 32) {
        v16bf ah = load_afrag(hi, abase, k0, half);
        v16bf al = load_afrag(lo, abase, k0, half);

        {   v16bf bh = load_bfrag(hi, bbase0, k0, half);
            v16bf bl = load_bfrag(lo, bbase0, k0, half);
            acc0 = __builtin_amdgcn_wmma_f32_16x16x32_bf16(false, ah, false, bh, (short)0, acc0, false, false);
            acc0 = __builtin_amdgcn_wmma_f32_16x16x32_bf16(false, ah, false, bl, (short)0, acc0, false, false);
            acc0 = __builtin_amdgcn_wmma_f32_16x16x32_bf16(false, al, false, bh, (short)0, acc0, false, false); }
        {   v16bf bh = load_bfrag(hi, bbase1, k0, half);
            v16bf bl = load_bfrag(lo, bbase1, k0, half);
            acc1 = __builtin_amdgcn_wmma_f32_16x16x32_bf16(false, ah, false, bh, (short)0, acc1, false, false);
            acc1 = __builtin_amdgcn_wmma_f32_16x16x32_bf16(false, ah, false, bl, (short)0, acc1, false, false);
            acc1 = __builtin_amdgcn_wmma_f32_16x16x32_bf16(false, al, false, bh, (short)0, acc1, false, false); }
        {   v16bf bh = load_bfrag(hi, bbase2, k0, half);
            v16bf bl = load_bfrag(lo, bbase2, k0, half);
            acc2 = __builtin_amdgcn_wmma_f32_16x16x32_bf16(false, ah, false, bh, (short)0, acc2, false, false);
            acc2 = __builtin_amdgcn_wmma_f32_16x16x32_bf16(false, ah, false, bl, (short)0, acc2, false, false);
            acc2 = __builtin_amdgcn_wmma_f32_16x16x32_bf16(false, al, false, bh, (short)0, acc2, false, false); }
        {   v16bf bh = load_bfrag(hi, bbase3, k0, half);
            v16bf bl = load_bfrag(lo, bbase3, k0, half);
            acc3 = __builtin_amdgcn_wmma_f32_16x16x32_bf16(false, ah, false, bh, (short)0, acc3, false, false);
            acc3 = __builtin_amdgcn_wmma_f32_16x16x32_bf16(false, ah, false, bl, (short)0, acc3, false, false);
            acc3 = __builtin_amdgcn_wmma_f32_16x16x32_bf16(false, al, false, bh, (short)0, acc3, false, false); }
    }

    // ---- epilogue: C layout lanes 0-15 -> M=p, lanes 16-31 -> M=p+8, N=lane&15
    const float inv_pos = dens[0];
    const float inv_neg = dens[1];

    AccF32 a0, a1, a2, a3;
    a0.v = acc0; a1.v = acc1; a2.v = acc2; a3.v = acc3;

    float local = 0.0f;
#pragma unroll
    for (int p = 0; p < 8; ++p) {
        const int   i  = m0 + p + half * 8;
        const float ri = r[i];
        const float si = s[i];
        const int   li = label[i];
#pragma unroll
        for (int t = 0; t < 4; ++t) {
            const int   j = n0 + t * 16 + lq;
            const float g = (t == 0) ? a0.f[p] : (t == 1) ? a1.f[p]
                          : (t == 2) ? a2.f[p] : a3.f[p];
            float sq = ri + r[j] - 2.0f * g
                     + 2.0f * PD_EPS * (si - s[j]) + DEPS2;
            float dist = sqrtf(fmaxf(sq, 1e-30f));
            if (li == label[j]) {
                local += dist * dist * inv_pos;
            } else {
                float h = fmaxf(1.0f - dist, 0.0f);
                local += h * h * inv_neg;
            }
        }
    }

#pragma unroll
    for (int off = 16; off > 0; off >>= 1) local += __shfl_xor(local, off, 32);
    if (lane == 0) wsum[wave] = local;
    __syncthreads();
    if (threadIdx.x == 0) {
        float tsum = 0.0f;
#pragma unroll
        for (int w = 0; w < 8; ++w) tsum += wsum[w];
        partials[blockIdx.y * 64 + blockIdx.x] = tsum;  // 128*64 = 8192 slots
    }
}

// ---------------------------------------------------------------------------
// Kernel 4: deterministic tree-reduce of 8192 partials -> scalar loss
// ---------------------------------------------------------------------------
__global__ __launch_bounds__(256) void reduce_kernel(
    const float* __restrict__ partials, float* __restrict__ out)
{
    __shared__ float sm[256];
    float v = 0.0f;
    for (int i = threadIdx.x; i < 8192; i += 256) v += partials[i];
    sm[threadIdx.x] = v;
    __syncthreads();
    for (int off = 128; off > 0; off >>= 1) {
        if (threadIdx.x < off) sm[threadIdx.x] += sm[threadIdx.x + off];
        __syncthreads();
    }
    if (threadIdx.x == 0) out[0] = sm[0];
}

// ---------------------------------------------------------------------------
// Host launcher
// ---------------------------------------------------------------------------
extern "C" void kernel_launch(void* const* d_in, const int* in_sizes, int n_in,
                              void* d_out, int out_size, void* d_ws, size_t ws_size,
                              hipStream_t stream) {
    (void)in_sizes; (void)n_in; (void)out_size; (void)ws_size;

    const float* emb   = (const float*)d_in[0];
    const int*   label = (const int*)d_in[1];
    float*       out   = (float*)d_out;

    char* ws = (char*)d_ws;
    const size_t HL = (size_t)N_ROWS * DIM * sizeof(bf16_t);   // 8 MB each
    bf16_t* hi       = (bf16_t*)(ws);
    bf16_t* lo       = (bf16_t*)(ws + HL);
    float*  r        = (float*) (ws + 2 * HL);
    float*  s        = (float*) (ws + 2 * HL + N_ROWS * sizeof(float));
    int*    cnt      = (int*)   (ws + 2 * HL + 2 * N_ROWS * sizeof(float));
    float*  dens     = (float*) (ws + 2 * HL + 2 * N_ROWS * sizeof(float) + 64);
    float*  partials = (float*) (ws + 2 * HL + 2 * N_ROWS * sizeof(float) + 128);

    init_kernel<<<1, 64, 0, stream>>>(cnt);
    rownorm_kernel<<<N_ROWS / 8, 256, 0, stream>>>(emb, label, hi, lo, r, s, cnt);
    denom_kernel<<<1, 32, 0, stream>>>(cnt, dens);
    gemm_loss_kernel<<<dim3(64, 128), 256, 0, stream>>>(hi, lo, r, s, label, dens, partials);
    reduce_kernel<<<1, 256, 0, stream>>>(partials, out);
}
